// Model_78769700209243
// MI455X (gfx1250) — matrix operand order repaired and checked by reading, not compiled, not run on hardware
//
#include <hip/hip_runtime.h>
#include <hip/hip_bf16.h>
#include <math.h>

typedef __attribute__((ext_vector_type(16))) _Float16 v16h;
typedef __attribute__((ext_vector_type(8)))  _Float16 v8h;
typedef __attribute__((ext_vector_type(8)))  float    v8f;

namespace {
constexpr int kB = 1024, kTO = 32, kTP = 30, kD = 2, kR = 256, kMAP = 10;
constexpr int kCF = 900;          // CONV_FLAT
constexpr int kKP = 928;          // CONV_FLAT padded to multiple of 32 for WMMA K-loop
constexpr int kNT = kB * kTO;     // frames per map stream (32768)
constexpr float kLAM = 0.0005f, kGAM = 0.1f;

// d_out layout (tuple flattened in return order)
constexpr int PE_OFF  = 0;                       // preds_enc  (B,TO,D)
constexpr int PD_OFF  = kB * kTO * kD;           // preds_dec  (B,TP,D)
constexpr int CPD_OFF = PD_OFF + kB * kTP * kD;  // cost_pos_dec scalar
constexpr int CR_OFF  = CPD_OFF + 1;             // cost_reward scalar

constexpr int kWROW = 40;  // LDS W-tile row pitch in halves (32 data + 8 pad)
} // namespace

// ---------------------------------------------------------------------------
// WMMA fragment loaders (16x16x32 f16, wave32; layouts per CDNA5 ISA 7.12.2)
// A (16x32): lanes 0-15 -> M=lane,   halves j<8: K=j,    j>=8: K=8+j
//            lanes16-31 -> M=lane-16 halves j<8: K=8+j,  j>=8: K=16+j
//   => per lane: two contiguous 8-half (16B) chunks at kb+hi*8 and kb+16+hi*8
// B (32x16): lane -> N=lane&15, same K order as A
//   => per lane: two 8-half chunks at kb+hi*8 and kb+16+hi*8
// ---------------------------------------------------------------------------
__device__ __forceinline__ v16h load_a_frag(const _Float16* A, int lda,
                                            int m_base, int kb, int lane) {
  const int hi = (lane >> 4) & 1;
  const int m  = m_base + (lane & 15);
  const _Float16* p = A + (size_t)m * lda + kb + hi * 8;
  v8h q0 = *reinterpret_cast<const v8h*>(p);
  v8h q1 = *reinterpret_cast<const v8h*>(p + 16);
  v16h r;
#pragma unroll
  for (int j = 0; j < 8; ++j) { r[j] = q0[j]; r[j + 8] = q1[j]; }
  return r;
}

// B fragment from the LDS-staged weight strip (row n_local, 32-K chunk)
__device__ __forceinline__ v16h load_b_lds(const _Float16* sWb, int tile, int lane) {
  const int hi = (lane >> 4) & 1;
  const int n_local = tile * 16 + (lane & 15);
  const _Float16* p = sWb + (size_t)n_local * kWROW + hi * 8;
  v8h q0 = *reinterpret_cast<const v8h*>(p);
  v8h q1 = *reinterpret_cast<const v8h*>(p + 16);
  v16h r;
#pragma unroll
  for (int j = 0; j < 8; ++j) { r[j] = q0[j]; r[j + 8] = q1[j]; }
  return r;
}

// ---------------------------------------------------------------------------
// Generic fused GEMM:  out = act( A0 @ W0^T [+ A1 @ W1^T] + bias0 [+ bias1] )
// A: (M,K) f16 row-major stride lda.  W: (N,K) f16 row-major (so W^T is KxN).
// One wave computes a 16(M) x 64(N) strip; block = 8 waves = 128 M rows.
// The 64-row W strip is shared by all 8 waves -> double-buffered LDS staging
// in 32-K chunks using GLOBAL_LOAD_ASYNC_TO_LDS_B128 (ASYNCcnt), one b128
// per thread per chunk; A fragments are software-pipelined one chunk ahead.
// REQUIRES: M % 128 == 0, N % 64 == 0, K % 32 == 0 (true at all call sites).
// ---------------------------------------------------------------------------
__global__ __launch_bounds__(256)
void gemm_wmma_kernel(const _Float16* __restrict__ A0, int lda0,
                      const _Float16* __restrict__ W0, int ldw0,
                      const _Float16* __restrict__ A1, int lda1,
                      const _Float16* __restrict__ W1, int ldw1,
                      const float* __restrict__ bias0,
                      const float* __restrict__ bias1,
                      float* __restrict__ outF, _Float16* __restrict__ outH,
                      int ldc, int M, int N, int K, int relu) {
  __shared__ _Float16 sW[2][64 * kWROW];

  const int tid  = threadIdx.x;
  const int wid  = tid >> 5;
  const int lane = tid & 31;
  const int hi   = (lane >> 4) & 1;
  const int m_base = blockIdx.x * 128 + wid * 16;
  const int n_blk  = blockIdx.y * 64;

  v8f acc[4] = {v8f{}, v8f{}, v8f{}, v8f{}};

  // Async stage of one 64x32 W chunk straight into LDS (no VGPR round trip):
  // thread t -> row t>>2, 8 halves (16B).  Tracked by ASYNCcnt.
  auto stage_async = [&](int buf, const _Float16* W, int ldw, int kb) {
    const int row = tid >> 2, part = tid & 3;
    const _Float16* src = W + (size_t)(n_blk + row) * ldw + kb + part * 8;
    // LDS byte offset = low 32 bits of the flat shared-memory address
    // (aperture mapping, CDNA5 ISA 10.2).
    unsigned ldsoff = (unsigned)(uintptr_t)(&sW[buf][row * kWROW + part * 8]);
    asm volatile("global_load_async_to_lds_b128 %0, %1, off"
                 :: "v"(ldsoff), "v"(src) : "memory");
  };

  auto run = [&](const _Float16* A, int lda, const _Float16* W, int ldw) {
    stage_async(0, W, ldw, 0);
    v16h a = load_a_frag(A, lda, m_base, 0, lane);   // overlap with staging
    asm volatile("s_wait_asynccnt 0x0" ::: "memory");
    __syncthreads();
    for (int kb = 0; kb < K; kb += 32) {
      const int cur = (kb >> 5) & 1;
      if (kb + 32 < K) stage_async(cur ^ 1, W, ldw, kb + 32);  // next chunk -> LDS
      v16h b0 = load_b_lds(sW[cur], 0, lane);
      v16h b1 = load_b_lds(sW[cur], 1, lane);
      v16h b2 = load_b_lds(sW[cur], 2, lane);
      v16h b3 = load_b_lds(sW[cur], 3, lane);
      v16h an = a;
      if (kb + 32 < K) an = load_a_frag(A, lda, m_base, kb + 32, lane);  // pipeline A
      asm volatile("s_wait_asynccnt 0x0" ::: "memory");  // own async stores in LDS
      __syncthreads();  // all waves staged next chunk; all reads of cur complete
      acc[0] = __builtin_amdgcn_wmma_f32_16x16x32_f16(false, a, false, b0, (short)0, acc[0], false, false);
      acc[1] = __builtin_amdgcn_wmma_f32_16x16x32_f16(false, a, false, b1, (short)0, acc[1], false, false);
      acc[2] = __builtin_amdgcn_wmma_f32_16x16x32_f16(false, a, false, b2, (short)0, acc[2], false, false);
      acc[3] = __builtin_amdgcn_wmma_f32_16x16x32_f16(false, a, false, b3, (short)0, acc[3], false, false);
      a = an;
    }
  };

  run(A0, lda0, W0, ldw0);
  if (A1 != nullptr) run(A1, lda1, W1, ldw1);

  // Epilogue: C/D layout -> VGPR r holds M=r (lanes 0-15) / M=8+r (lanes 16-31)
  const int col_l = lane & 15;
#pragma unroll
  for (int i = 0; i < 4; ++i) {
    const int col = n_blk + 16 * i + col_l;
    float bv = 0.f;
    if (bias0) bv += bias0[col];
    if (bias1) bv += bias1[col];
#pragma unroll
    for (int r = 0; r < 8; ++r) {
      const int row = m_base + r + hi * 8;
      float v = acc[i][r] + bv;
      if (relu) v = fmaxf(v, 0.f);
      const size_t o = (size_t)row * ldc + col;
      if (outF) outF[o] = v;
      if (outH) outH[o] = (_Float16)v;
    }
  }
}

// ---------------------------------------------------------------------------
// 3-layer conv stack per 10x10 map frame, all intermediates in LDS.
// blockIdx.x in [0, 2*NT): first NT frames = gt maps, rest = policy maps.
// Output f16 rows of length kKP (tail zero-padded) in NCHW-flat order.
// ---------------------------------------------------------------------------
__global__ __launch_bounds__(128)
void conv_stack_kernel(const float* __restrict__ gt_map,
                       const float* __restrict__ pol_map,
                       const float* __restrict__ c1w, const float* __restrict__ c1b,
                       const float* __restrict__ c2w, const float* __restrict__ c2b,
                       const float* __restrict__ c3w, const float* __restrict__ c3b,
                       _Float16* __restrict__ out_gt, _Float16* __restrict__ out_pol) {
  const int img = blockIdx.x;
  const bool pol = img >= kNT;
  const int idx = pol ? img - kNT : img;
  const float* src = (pol ? pol_map : gt_map) + (size_t)idx * (kMAP * kMAP * 3);
  _Float16* dst = (pol ? out_pol : out_gt) + (size_t)idx * kKP;

  __shared__ float s_in[3][kMAP][kMAP];
  __shared__ float s_c1[6][kMAP][kMAP];
  __shared__ float s_c2[9][kMAP][kMAP];
  __shared__ float s_w1[450 + 6];
  __shared__ float s_w2[486 + 9];
  __shared__ float s_w3[729 + 9];

  const int tid = threadIdx.x;
  for (int i = tid; i < 456; i += 128) s_w1[i] = (i < 450) ? c1w[i] : c1b[i - 450];
  for (int i = tid; i < 495; i += 128) s_w2[i] = (i < 486) ? c2w[i] : c2b[i - 486];
  for (int i = tid; i < 738; i += 128) s_w3[i] = (i < 729) ? c3w[i] : c3b[i - 729];
  for (int i = tid; i < 300; i += 128) {       // HWC -> CHW
    int c = i % 3, xy = i / 3;
    s_in[c][xy / kMAP][xy % kMAP] = src[i];
  }
  __syncthreads();

  // conv1: 3->6 ch, 5x5 SAME
  for (int o = tid; o < 600; o += 128) {
    int oc = o / 100, y = (o % 100) / 10, x = o % 10;
    float s = s_w1[450 + oc];
    for (int ic = 0; ic < 3; ++ic)
      for (int ky = 0; ky < 5; ++ky) {
        int yy = y + ky - 2; if (yy < 0 || yy > 9) continue;
        for (int kx = 0; kx < 5; ++kx) {
          int xx = x + kx - 2; if (xx < 0 || xx > 9) continue;
          s += s_in[ic][yy][xx] * s_w1[((oc * 3 + ic) * 5 + ky) * 5 + kx];
        }
      }
    s_c1[oc][y][x] = fmaxf(s, 0.f);
  }
  __syncthreads();

  // conv2: 6->9 ch, 3x3 SAME
  for (int o = tid; o < 900; o += 128) {
    int oc = o / 100, y = (o % 100) / 10, x = o % 10;
    float s = s_w2[486 + oc];
    for (int ic = 0; ic < 6; ++ic)
      for (int ky = 0; ky < 3; ++ky) {
        int yy = y + ky - 1; if (yy < 0 || yy > 9) continue;
        for (int kx = 0; kx < 3; ++kx) {
          int xx = x + kx - 1; if (xx < 0 || xx > 9) continue;
          s += s_c1[ic][yy][xx] * s_w2[((oc * 6 + ic) * 3 + ky) * 3 + kx];
        }
      }
    s_c2[oc][y][x] = fmaxf(s, 0.f);
  }
  __syncthreads();

  // conv3: 9->9 ch, 3x3 SAME -> f16 out
  for (int o = tid; o < 900; o += 128) {
    int oc = o / 100, y = (o % 100) / 10, x = o % 10;
    float s = s_w3[729 + oc];
    for (int ic = 0; ic < 9; ++ic)
      for (int ky = 0; ky < 3; ++ky) {
        int yy = y + ky - 1; if (yy < 0 || yy > 9) continue;
        for (int kx = 0; kx < 3; ++kx) {
          int xx = x + kx - 1; if (xx < 0 || xx > 9) continue;
          s += s_c2[ic][yy][xx] * s_w3[((oc * 9 + ic) * 3 + ky) * 3 + kx];
        }
      }
    dst[o] = (_Float16)fmaxf(s, 0.f);
  }
  for (int k = kCF + tid; k < kKP; k += 128) dst[k] = (_Float16)0.f;  // zero pad
}

// --------------------------- small elementwise kernels ----------------------
__global__ void zero_f32_kernel(float* p, int n) {
  int i = blockIdx.x * blockDim.x + threadIdx.x; if (i < n) p[i] = 0.f;
}
__global__ void zero_f16_kernel(_Float16* p, int n) {
  int i = blockIdx.x * blockDim.x + threadIdx.x; if (i < n) p[i] = (_Float16)0.f;
}
__global__ void cvt_f16_kernel(const float* s, _Float16* d, int n) {
  int i = blockIdx.x * blockDim.x + threadIdx.x; if (i < n) d[i] = (_Float16)s[i];
}
__global__ void cvt_wc_pad_kernel(const float* s, _Float16* d) {  // (128,900)->(128,928)
  int i = blockIdx.x * blockDim.x + threadIdx.x;
  if (i >= 128 * kKP) return;
  int n = i / kKP, k = i % kKP;
  d[i] = (k < kCF) ? (_Float16)s[n * kCF + k] : (_Float16)0.f;
}

// emb_pose: relu(traj @ we_w^T + we_b), K=2 -> E1 cols [0,128)
__global__ void emb_pose_kernel(const float* __restrict__ traj,
                                const float* __restrict__ we_w,
                                const float* __restrict__ we_b,
                                _Float16* __restrict__ E1) {
  int i = blockIdx.x * blockDim.x + threadIdx.x;
  if (i >= kNT * (kR / 2)) return;
  int row = i / (kR / 2), j = i % (kR / 2);
  float v = traj[row * 2] * we_w[j * 2] + traj[row * 2 + 1] * we_w[j * 2 + 1] + we_b[j];
  E1[(size_t)row * kR + j] = (_Float16)fmaxf(v, 0.f);
}

// LSTM gates/cell update.  g layout (B, 4R): [i | f | g | o]
__global__ void lstm_cell_kernel(const float* __restrict__ g,
                                 float* __restrict__ c,
                                 float* __restrict__ h32,
                                 _Float16* __restrict__ h16) {
  int i = blockIdx.x * blockDim.x + threadIdx.x;
  if (i >= kB * kR) return;
  int b = i >> 8, r = i & (kR - 1);
  const float* gr = g + (size_t)b * 4 * kR;
  float ig = 1.f / (1.f + expf(-gr[r]));
  float fg = 1.f / (1.f + expf(-gr[kR + r]));
  float gg = tanhf(gr[2 * kR + r]);
  float og = 1.f / (1.f + expf(-gr[3 * kR + r]));
  float cn = fg * c[i] + ig * gg;
  c[i] = cn;
  float hn = og * tanhf(cn);
  h32[i] = hn; h16[i] = (_Float16)hn;
}

// pred = h @ W^T + b, W:(D,R).  outbase[b*bstride + d]
__global__ void proj2_kernel(const float* __restrict__ h,
                             const float* __restrict__ W,
                             const float* __restrict__ bias,
                             float* __restrict__ outbase, int bstride) {
  int i = blockIdx.x * blockDim.x + threadIdx.x;
  if (i >= kB * kD) return;
  int b = i >> 1, d = i & 1;
  const float* hr = h + (size_t)b * kR;
  const float* wr = W + (size_t)d * kR;
  float s = bias[d];
  for (int r = 0; r < kR; ++r) s += hr[r] * wr[r];
  outbase[(size_t)b * bstride + d] = s;
}

// e = relu(pred @ wd_w^T + wd_b), wd_w:(R,D), K=2 -> f16
__global__ void dec_embed_kernel(const float* __restrict__ predbase, int bstride,
                                 const float* __restrict__ wd_w,
                                 const float* __restrict__ wd_b,
                                 _Float16* __restrict__ e16) {
  int i = blockIdx.x * blockDim.x + threadIdx.x;
  if (i >= kB * kR) return;
  int b = i >> 8, r = i & (kR - 1);
  float p0 = predbase[(size_t)b * bstride];
  float p1 = predbase[(size_t)b * bstride + 1];
  float v = p0 * wd_w[r * 2] + p1 * wd_w[r * 2 + 1] + wd_b[r];
  e16[i] = (_Float16)fmaxf(v, 0.f);
}

// per-(b,t) reward dot: [conv(900)|pose(2)] . fr_w + fr_b
__global__ void reward_dot_kernel(const _Float16* __restrict__ conv,
                                  const float* __restrict__ pose, int pose_bstride,
                                  int pose_toff,
                                  const float* __restrict__ fr_w,
                                  const float* __restrict__ fr_b,
                                  float* __restrict__ rbuf) {
  int i = blockIdx.x * blockDim.x + threadIdx.x;
  if (i >= kB * (kTO - 2)) return;
  int b = i / (kTO - 2), j = i % (kTO - 2);
  const _Float16* cr = conv + (size_t)(b * kTO + j + 2) * kKP;
  float s = fr_b[0];
  for (int k = 0; k < kCF; ++k) s += (float)cr[k] * fr_w[k];
  const float* pr = pose + (size_t)b * pose_bstride + (size_t)(j + pose_toff) * kD;
  s += pr[0] * fr_w[kCF] + pr[1] * fr_w[kCF + 1];
  rbuf[i] = s;
}

// acc[0] += sum_b -log(mean(rgt)-mean(rpol)+1+1e-20)
__global__ void reward_cost_kernel(const float* __restrict__ rg,
                                   const float* __restrict__ rp,
                                   float* __restrict__ acc) {
  int b = blockIdx.x * blockDim.x + threadIdx.x;
  if (b >= kB) return;
  float sg = 0.f, sp = 0.f;
  for (int j = 0; j < kTO - 2; ++j) {
    sg += rg[b * (kTO - 2) + j];
    sp += rp[b * (kTO - 2) + j];
  }
  float diff = sg / (kTO - 2) - sp / (kTO - 2) + 1.0f + 1e-20f;
  atomicAdd(acc, -logf(diff));
}

__global__ void sumsq_kernel(const float* __restrict__ x, int n, float* acc) {
  __shared__ float s[256];
  float v = 0.f;
  for (int i = blockIdx.x * blockDim.x + threadIdx.x; i < n; i += gridDim.x * blockDim.x)
    v += x[i] * x[i];
  s[threadIdx.x] = v; __syncthreads();
  for (int st = 128; st > 0; st >>= 1) {
    if ((int)threadIdx.x < st) s[threadIdx.x] += s[threadIdx.x + st];
    __syncthreads();
  }
  if (threadIdx.x == 0) atomicAdd(acc, s[0]);
}

__global__ void sqdiff_kernel(const float* __restrict__ a, const float* __restrict__ b,
                              int n, float* acc) {
  __shared__ float s[256];
  float v = 0.f;
  for (int i = blockIdx.x * blockDim.x + threadIdx.x; i < n; i += gridDim.x * blockDim.x) {
    float d = a[i] - b[i]; v += d * d;
  }
  s[threadIdx.x] = v; __syncthreads();
  for (int st = 128; st > 0; st >>= 1) {
    if ((int)threadIdx.x < st) s[threadIdx.x] += s[threadIdx.x + st];
    __syncthreads();
  }
  if (threadIdx.x == 0) atomicAdd(acc, s[0]);
}

// acc: [0]=sum(-log diff) [1]=sum sq pos [2]=l2 conv [3]=l2 reward [4]=l2 pose
__global__ void finalize_kernel(const float* __restrict__ acc, float* __restrict__ out) {
  if (threadIdx.x != 0 || blockIdx.x != 0) return;
  float cost_reward = acc[0] / (float)kB;
  float cost_policy = -acc[0] / (float)kB;
  float l2c = kLAM * acc[2], l2r = kLAM * acc[3], l2p = kLAM * acc[4];
  out[CPD_OFF] = acc[1] / (float)(kB * kTP) + kGAM * cost_policy + l2p + l2c;
  out[CR_OFF]  = cost_reward + l2r + l2c;
}

// ---------------------------------------------------------------------------
static void launch_gemm(hipStream_t s,
                        const _Float16* A0, int lda0, const _Float16* W0, int ldw0,
                        const _Float16* A1, int lda1, const _Float16* W1, int ldw1,
                        const float* b0, const float* b1,
                        float* oF, _Float16* oH, int ldc, int M, int N, int K, int relu) {
  dim3 grid((M + 127) / 128, (N + 63) / 64);
  gemm_wmma_kernel<<<grid, 256, 0, s>>>(A0, lda0, W0, ldw0, A1, lda1, W1, ldw1,
                                        b0, b1, oF, oH, ldc, M, N, K, relu);
}

extern "C" void kernel_launch(void* const* d_in, const int* in_sizes, int n_in,
                              void* d_out_v, int out_size, void* d_ws, size_t ws_size,
                              hipStream_t stream) {
  (void)n_in; (void)out_size; (void)ws_size;
  const float* gt_traj_enc = (const float*)d_in[0];
  const float* gt_map      = (const float*)d_in[1];
  const float* pol_map     = (const float*)d_in[2];
  const float* gt_traj_dec = (const float*)d_in[3];
  const float* conv1_w = (const float*)d_in[4];
  const float* conv1_b = (const float*)d_in[5];
  const float* conv2_w = (const float*)d_in[6];
  const float* conv2_b = (const float*)d_in[7];
  const float* conv3_w = (const float*)d_in[8];
  const float* conv3_b = (const float*)d_in[9];
  const float* enc_Wih = (const float*)d_in[10];
  const float* enc_Whh = (const float*)d_in[11];
  const float* enc_bih = (const float*)d_in[12];
  const float* enc_bhh = (const float*)d_in[13];
  const float* dec_Wih = (const float*)d_in[14];
  const float* dec_Whh = (const float*)d_in[15];
  const float* dec_bih = (const float*)d_in[16];
  const float* dec_bhh = (const float*)d_in[17];
  const float* we_w = (const float*)d_in[18];
  const float* we_b = (const float*)d_in[19];
  const float* wc_w = (const float*)d_in[20];
  const float* wc_b = (const float*)d_in[21];
  const float* wcc_w = (const float*)d_in[22];
  const float* wcc_b = (const float*)d_in[23];
  const float* wd_w = (const float*)d_in[24];
  const float* wd_b = (const float*)d_in[25];
  const float* owe_w = (const float*)d_in[26];
  const float* owe_b = (const float*)d_in[27];
  const float* owd_w = (const float*)d_in[28];
  const float* owd_b = (const float*)d_in[29];
  const float* fr_w = (const float*)d_in[30];
  const float* fr_b = (const float*)d_in[31];
  float* out = (float*)d_out_v;

  // ---- workspace bump allocator (256B aligned) ----
  uintptr_t base = (uintptr_t)d_ws;
  size_t off = 0;
  auto alloc = [&](size_t bytes) -> void* {
    void* p = (void*)(base + off);
    off = (off + bytes + 255) & ~(size_t)255;
    return p;
  };
  _Float16* encWih_h = (_Float16*)alloc((size_t)4 * kR * kR * 2);
  _Float16* encWhh_h = (_Float16*)alloc((size_t)4 * kR * kR * 2);
  _Float16* decWih_h = (_Float16*)alloc((size_t)4 * kR * kR * 2);
  _Float16* decWhh_h = (_Float16*)alloc((size_t)4 * kR * kR * 2);
  _Float16* wcc_h    = (_Float16*)alloc((size_t)kR * kR * 2);
  _Float16* wc_h     = (_Float16*)alloc((size_t)128 * kKP * 2);
  _Float16* conv_gt_h  = (_Float16*)alloc((size_t)kNT * kKP * 2);
  _Float16* conv_pol_h = (_Float16*)alloc((size_t)kNT * kKP * 2);
  _Float16* E1     = (_Float16*)alloc((size_t)kNT * kR * 2);
  _Float16* emb_h  = (_Float16*)alloc((size_t)kNT * kR * 2);
  float*    gbuf   = (float*)alloc((size_t)kB * 4 * kR * 4);
  float*    h32    = (float*)alloc((size_t)kB * kR * 4);
  float*    c32    = (float*)alloc((size_t)kB * kR * 4);
  _Float16* h16    = (_Float16*)alloc((size_t)kB * kR * 2);
  _Float16* e16    = (_Float16*)alloc((size_t)kB * kR * 2);
  float*    rbuf_g = (float*)alloc((size_t)kB * (kTO - 2) * 4);
  float*    rbuf_p = (float*)alloc((size_t)kB * (kTO - 2) * 4);
  float*    acc    = (float*)alloc(8 * 4);

  auto nb = [](int n, int t) { return (n + t - 1) / t; };

  // 0) zero accumulators
  zero_f32_kernel<<<1, 8, 0, stream>>>(acc, 8);

  // 1) convert weights to f16
  cvt_f16_kernel<<<nb(4 * kR * kR, 256), 256, 0, stream>>>(enc_Wih, encWih_h, 4 * kR * kR);
  cvt_f16_kernel<<<nb(4 * kR * kR, 256), 256, 0, stream>>>(enc_Whh, encWhh_h, 4 * kR * kR);
  cvt_f16_kernel<<<nb(4 * kR * kR, 256), 256, 0, stream>>>(dec_Wih, decWih_h, 4 * kR * kR);
  cvt_f16_kernel<<<nb(4 * kR * kR, 256), 256, 0, stream>>>(dec_Whh, decWhh_h, 4 * kR * kR);
  cvt_f16_kernel<<<nb(kR * kR, 256), 256, 0, stream>>>(wcc_w, wcc_h, kR * kR);
  cvt_wc_pad_kernel<<<nb(128 * kKP, 256), 256, 0, stream>>>(wc_w, wc_h);

  // 2) conv stacks for both map streams (f16 rows, padded to kKP)
  conv_stack_kernel<<<2 * kNT, 128, 0, stream>>>(gt_map, pol_map,
      conv1_w, conv1_b, conv2_w, conv2_b, conv3_w, conv3_b, conv_gt_h, conv_pol_h);

  // 3) emb_pose -> E1[:, 0:128)
  emb_pose_kernel<<<nb(kNT * 128, 256), 256, 0, stream>>>(gt_traj_enc, we_w, we_b, E1);

  // 4) emb_conv = relu(conv_gt @ wc_w^T + wc_b) -> E1[:, 128:256)   (WMMA)
  launch_gemm(stream, conv_gt_h, kKP, wc_h, kKP, nullptr, 0, nullptr, 0,
              wc_b, nullptr, nullptr, E1 + 128, kR, kNT, 128, kKP, 1);

  // 5) emb = relu(E1 @ wcc_w^T + wcc_b) -> emb_h                    (WMMA)
  launch_gemm(stream, E1, kR, wcc_h, kR, nullptr, 0, nullptr, 0,
              wcc_b, nullptr, nullptr, emb_h, kR, kNT, kR, kR, 1);

  // 6) LSTM encoder
  zero_f32_kernel<<<nb(kB * kR, 256), 256, 0, stream>>>(h32, kB * kR);
  zero_f32_kernel<<<nb(kB * kR, 256), 256, 0, stream>>>(c32, kB * kR);
  zero_f16_kernel<<<nb(kB * kR, 256), 256, 0, stream>>>(h16, kB * kR);
  for (int t = 0; t < kTO; ++t) {
    // g = x_t @ Wih^T + h @ Whh^T + bih + bhh                       (WMMA x2)
    launch_gemm(stream, emb_h + (size_t)t * kR, kTO * kR, encWih_h, kR,
                h16, kR, encWhh_h, kR, enc_bih, enc_bhh,
                gbuf, nullptr, 4 * kR, kB, 4 * kR, kR, 0);
    lstm_cell_kernel<<<nb(kB * kR, 256), 256, 0, stream>>>(gbuf, c32, h32, h16);
    proj2_kernel<<<nb(kB * kD, 256), 256, 0, stream>>>(h32, owe_w, owe_b,
        out + PE_OFF + t * kD, kTO * kD);
  }

  // 7) rewards
  reward_dot_kernel<<<nb(kB * (kTO - 2), 128), 128, 0, stream>>>(
      conv_gt_h, gt_traj_enc, kTO * kD, 2, fr_w, fr_b, rbuf_g);
  reward_dot_kernel<<<nb(kB * (kTO - 2), 128), 128, 0, stream>>>(
      conv_pol_h, out + PE_OFF, kTO * kD, 1, fr_w, fr_b, rbuf_p);
  reward_cost_kernel<<<nb(kB, 256), 256, 0, stream>>>(rbuf_g, rbuf_p, acc + 0);

  // 8) LSTM decoder
  zero_f32_kernel<<<nb(kB * kR, 256), 256, 0, stream>>>(h32, kB * kR);
  zero_f32_kernel<<<nb(kB * kR, 256), 256, 0, stream>>>(c32, kB * kR);
  zero_f16_kernel<<<nb(kB * kR, 256), 256, 0, stream>>>(h16, kB * kR);
  dec_embed_kernel<<<nb(kB * kR, 256), 256, 0, stream>>>(
      gt_traj_enc + (kTO - 1) * kD, kTO * kD, wd_w, wd_b, e16);  // emb0 from last pose
  for (int t = 0; t < kTP; ++t) {
    launch_gemm(stream, e16, kR, decWih_h, kR, h16, kR, decWhh_h, kR,
                dec_bih, dec_bhh, gbuf, nullptr, 4 * kR, kB, 4 * kR, kR, 0);
    lstm_cell_kernel<<<nb(kB * kR, 256), 256, 0, stream>>>(gbuf, c32, h32, h16);
    proj2_kernel<<<nb(kB * kD, 256), 256, 0, stream>>>(h32, owd_w, owd_b,
        out + PD_OFF + t * kD, kTP * kD);
    dec_embed_kernel<<<nb(kB * kR, 256), 256, 0, stream>>>(
        out + PD_OFF + t * kD, kTP * kD, wd_w, wd_b, e16);
  }

  // 9) cost_pos_dec raw sum
  sqdiff_kernel<<<64, 256, 0, stream>>>(out + PD_OFF, gt_traj_dec, kB * kTP * kD, acc + 1);

  // 10) L2 regularizers
  const int conv_idx[6] = {4, 5, 6, 7, 8, 9};
  for (int i = 0; i < 6; ++i)
    sumsq_kernel<<<4, 256, 0, stream>>>((const float*)d_in[conv_idx[i]],
                                        in_sizes[conv_idx[i]], acc + 2);
  const int rw_idx[2] = {30, 31};
  for (int i = 0; i < 2; ++i)
    sumsq_kernel<<<4, 256, 0, stream>>>((const float*)d_in[rw_idx[i]],
                                        in_sizes[rw_idx[i]], acc + 3);
  const int pose_idx[12] = {18, 19, 20, 21, 22, 23, 24, 25, 26, 27, 28, 29};
  for (int i = 0; i < 12; ++i)
    sumsq_kernel<<<32, 256, 0, stream>>>((const float*)d_in[pose_idx[i]],
                                         in_sizes[pose_idx[i]], acc + 4);

  // 11) finalize scalar outputs
  finalize_kernel<<<1, 1, 0, stream>>>(acc, out);
}
